// TV2D_2757369004816
// MI455X (gfx1250) — compile-verified, matrix operand order checked by reading, not run on hardware
//
#include <hip/hip_runtime.h>

// TV2D prox (anisotropic, dual projected gradient) on MI455X / gfx1250.
// One wave32 per TWO 14x14 maps (interleaved chains hide WMMA->VALU hazards).
//
// Per map-iteration: 4 WMMA total, via layout dualities:
//   pT  = (-Gh)*sh + uv^T*(-Gvt) + x^T  = wmma_f16(Bcomb, Acomb, xTh)
//   sh += tau*G^T * pT  (f16 acc+clip)  = wmma_f16(TVC, [pT|und], sh)
//   uv += tau*G   * p   (f16 acc+clip)  = wmma_f16(TVC, [p_f16|und], uv)
//   p   = uh*(-Gh^T) + (-Gvt^T)*uv + x  = wmma_f32(Acomb, Bcomb, x)
// where Acomb = [sh_f16 | C2-const], Bcomb = [C1-const | uv_f16] (K=32 fusion).
// A packed f16 C/D tile as the A operand is the transpose of itself as B, and
// the constants satisfy the same duality (C1-as-A == -Gh, C2-as-B == -Gvt),
// so swapping operand roles yields the transposed p-equation for free.
// ppq/pTq upper halves are undef: the consuming A-constant has zero K=16..31.

typedef __attribute__((ext_vector_type(16))) _Float16 v16h;
typedef __attribute__((ext_vector_type(8)))  _Float16 v8h;
typedef __attribute__((ext_vector_type(8)))  float    v8f;
typedef __attribute__((ext_vector_type(4)))  unsigned v4u;

#define TV_HW     14
#define TV_LAM    0.005f     // alpha/2 = 0.01/2
#define TV_TAU    0.125f
#define TV_ITERS  200

#if __has_builtin(__builtin_amdgcn_wmma_f16_16x16x32_f16)
#define TV_F16ACC 1
#else
#define TV_F16ACC 0
#endif

#define TV_WMMA_F32(A, B, C) \
  __builtin_amdgcn_wmma_f32_16x16x32_f16(false, (A), false, (B), (short)0, (C), false, false)
#if TV_F16ACC
#define TV_WMMA_F16(A, B, C) \
  __builtin_amdgcn_wmma_f16_16x16x32_f16(false, (A), false, (B), (short)0, (C), false, false)
#endif

// Pack an f32 C/D tile into an 8-half vector (f16 C/D / operand-half form).
__device__ __forceinline__ v8h tv_pack8(v8f s) {
  v8h r;
#pragma unroll
  for (int v = 0; v < 4; ++v) {
    auto pk = __builtin_amdgcn_cvt_pkrtz(s[2 * v], s[2 * v + 1]);
    r[2 * v]     = pk[0];
    r[2 * v + 1] = pk[1];
  }
  return r;
}
// Pack an f32 tile into elements 0..7 (K=0..15) of a tuple (high untouched).
__device__ __forceinline__ void tv_pack_low(v16h& r, v8f s) {
#pragma unroll
  for (int v = 0; v < 4; ++v) {
    auto pk = __builtin_amdgcn_cvt_pkrtz(s[2 * v], s[2 * v + 1]);
    r[2 * v]     = pk[0];
    r[2 * v + 1] = pk[1];
  }
}
// Pack an f32 tile into elements 8..15 (K=16..31) of a tuple.
__device__ __forceinline__ void tv_pack_high(v16h& r, v8f s) {
#pragma unroll
  for (int v = 0; v < 4; ++v) {
    auto pk = __builtin_amdgcn_cvt_pkrtz(s[2 * v], s[2 * v + 1]);
    r[8 + 2 * v]     = pk[0];
    r[8 + 2 * v + 1] = pk[1];
  }
}

__device__ __forceinline__ v8h tv_low(v16h a) {
  return __builtin_shufflevector(a, a, 0, 1, 2, 3, 4, 5, 6, 7);
}
__device__ __forceinline__ v8h tv_high(v16h a) {
  return __builtin_shufflevector(a, a, 8, 9, 10, 11, 12, 13, 14, 15);
}
__device__ __forceinline__ v16h tv_setlow(v16h a, v8h lo) {
#pragma unroll
  for (int e = 0; e < 8; ++e) a[e] = lo[e];
  return a;
}
__device__ __forceinline__ v16h tv_sethigh(v16h a, v8h hi) {
#pragma unroll
  for (int e = 0; e < 8; ++e) a[8 + e] = hi[e];
  return a;
}

#if TV_F16ACC
// Clamp 8 packed halves to [-0.005, 0.005] in place: 2 v_pk ops per VGPR.
// 0x1d1f == (_Float16)0.005f, 0x9d1f == negated.
__device__ __forceinline__ v8h tv_clip8(v8h u) {
  v4u b = __builtin_bit_cast(v4u, u);
#pragma unroll
  for (int e = 0; e < 4; ++e) {
    asm("v_pk_max_num_f16 %0, 0x9d1f, %0 op_sel_hi:[0,1]\n\t"
        "v_pk_min_num_f16 %0, 0x1d1f, %0 op_sel_hi:[0,1]"
        : "+v"(b[e]));
  }
  return __builtin_bit_cast(v8h, b);
}
#endif

// Constant tile, low half (K=0..15). g(k, n): k = K (B-row / A-col), n = lane%16.
template <typename F>
__device__ __forceinline__ v16h tv_const_lo(int lane, F g) {
  v16h r = {};
  const int n  = lane & 15;
  const int hb = (lane >> 4) & 1;
#pragma unroll
  for (int v = 0; v < 4; ++v) {
    const int k0 = 2 * v + 8 * hb;
    r[2 * v]     = (_Float16)g(k0, n);
    r[2 * v + 1] = (_Float16)g(k0 + 1, n);
  }
  return r;
}
// Constant tile, high half (K=16..31); g receives k-16 in 0..15.
template <typename F>
__device__ __forceinline__ v16h tv_const_hi(int lane, F g) {
  v16h r = {};
  const int n  = lane & 15;
  const int hb = (lane >> 4) & 1;
#pragma unroll
  for (int v = 0; v < 4; ++v) {
    const int k0 = 2 * v + 8 * hb;
    r[8 + 2 * v]     = (_Float16)g(k0, n);
    r[8 + 2 * v + 1] = (_Float16)g(k0 + 1, n);
  }
  return r;
}

struct TvMap {
  v8f  x;              // f32 input tile
  v8f  p;              // loop-carried primal (p_0 = x)
  v8h  xTh;            // packed x^T (f16), C operand of the pT WMMA
  v16h Acomb;          // [sh_f16 | C2-const]
  v16h Bcomb;          // [C1-const | uv_f16]
  v8f  xT, pTf, shf, uvf;  // fallback path only
};

__global__ __launch_bounds__(256) void tv2d_prox_kernel(const float* __restrict__ X,
                                                        float* __restrict__ OUT,
                                                        int nmaps) {
  const int lane = threadIdx.x & 31;
  const int wave = threadIdx.x >> 5;
  const long long pair = (long long)blockIdx.x * (blockDim.x >> 5) + wave;
  const long long m0   = pair * 2;
  if (m0 >= nmaps) return;  // wave-uniform

  const int n  = lane & 15;
  const int hb = (lane >> 4) & 1;

  // ---- constant operator tiles ----
  // C1[j][c] = (c==j) - (c==j+1)  (-Gh^T as B; as A it is -Gh), j<13, c<14.
  const v16h C1lo = tv_const_lo(lane, [](int j, int c) {
    float r = 0.f;
    if (j < 13 && c < 14) { if (c == j) r += 1.f; if (c == j + 1) r -= 1.f; }
    return r;
  });
  // C2[i][r] = (i==r) - (i==r+1)  (-Gvt^T as A; as B it is -Gvt), r<13, i<14.
  const v16h C2hi = tv_const_hi(lane, [](int r, int i) {
    float v = 0.f;
    if (r < 13 && i < 14) { if (i == r) v += 1.f; if (i == r + 1) v -= 1.f; }
    return v;
  });
  // TVC[M=j][K=c] = tau*((c==j+1) - (c==j)): both tau*Gh^T and tau*Gvt as A.
  const v16h TVC = tv_const_lo(lane, [](int c, int j) {
    float v = 0.f;
    if (j < 13 && c < 14) { if (c == j + 1) v += TV_TAU; if (c == j) v -= TV_TAU; }
    return v;
  });

  // ---- per-map state ----
  TvMap s[2];
#pragma unroll
  for (int m = 0; m < 2; ++m) {
    s[m].x = {}; s[m].xT = {};
    s[m].shf = {}; s[m].uvf = {};
    s[m].Acomb = C2hi;   // sh = 0 in low half
    s[m].Bcomb = C1lo;   // uv = 0 in high half
    const long long map = m0 + m;
    if (map < nmaps) {
      const float* xp = X + map * (TV_HW * TV_HW);
#pragma unroll
      for (int r = 0; r < 8; ++r) {
        const int row = r + 8 * hb;
        const bool ok = (row < TV_HW && n < TV_HW);
        s[m].x[r]  = ok ? xp[row * TV_HW + n] : 0.f;
        s[m].xT[r] = ok ? xp[n * TV_HW + row] : 0.f;
      }
    }
    s[m].p   = s[m].x;            // p_0 = x - div(0)
    s[m].xTh = tv_pack8(s[m].xT); // f16 x^T (C of the pT WMMA)
    s[m].pTf = s[m].xT;           // fallback
  }

#pragma unroll 1
  for (int it = 0; it < TV_ITERS; ++it) {
#if TV_F16ACC
    // pack p; upper half undef (annihilated by TVC's zero K=16..31 half)
    v16h ppq[2];
#pragma unroll
    for (int m = 0; m < 2; ++m) tv_pack_low(ppq[m], s[m].p);
    // pT from current duals, directly in f16 operand form (swapped roles)
    v8h pTh[2];
#pragma unroll
    for (int m = 0; m < 2; ++m) pTh[m] = TV_WMMA_F16(s[m].Bcomb, s[m].Acomb, s[m].xTh);
    // dual ascent + projection; duals stay in f16 operand form in place
#pragma unroll
    for (int m = 0; m < 2; ++m) {
      v16h pTq;  // upper half undef
      pTq = tv_setlow(pTq, pTh[m]);
      s[m].Acomb = tv_setlow(s[m].Acomb, tv_clip8(TV_WMMA_F16(TVC, pTq, tv_low(s[m].Acomb))));
    }
#pragma unroll
    for (int m = 0; m < 2; ++m)
      s[m].Bcomb = tv_sethigh(s[m].Bcomb, tv_clip8(TV_WMMA_F16(TVC, ppq[m], tv_high(s[m].Bcomb))));
#else
    // f32-accumulate fallback (also keeps the host pass semantically valid)
    v16h ppq[2], pTq[2];
#pragma unroll
    for (int m = 0; m < 2; ++m) { tv_pack_low(ppq[m], s[m].p); tv_pack_low(pTq[m], s[m].pTf); }
#pragma unroll
    for (int m = 0; m < 2; ++m) {
      s[m].shf = TV_WMMA_F32(TVC, pTq[m], s[m].shf);
      s[m].uvf = TV_WMMA_F32(TVC, ppq[m], s[m].uvf);
#pragma unroll
      for (int r = 0; r < 8; ++r) {
        s[m].shf[r] = __builtin_amdgcn_fmed3f(s[m].shf[r], -TV_LAM, TV_LAM);
        s[m].uvf[r] = __builtin_amdgcn_fmed3f(s[m].uvf[r], -TV_LAM, TV_LAM);
      }
      tv_pack_low(s[m].Acomb, s[m].shf);
      tv_pack_high(s[m].Bcomb, s[m].uvf);
    }
#pragma unroll
    for (int m = 0; m < 2; ++m) s[m].pTf = TV_WMMA_F32(s[m].Bcomb, s[m].Acomb, s[m].xT);
#endif

    // fused primal update from the new duals (single K=32 WMMA, f32 out)
#pragma unroll
    for (int m = 0; m < 2; ++m) s[m].p = TV_WMMA_F32(s[m].Acomb, s[m].Bcomb, s[m].x);
  }

#pragma unroll
  for (int m = 0; m < 2; ++m) {
    const long long map = m0 + m;
    if (map < nmaps) {
      float* op = OUT + map * (TV_HW * TV_HW);
#pragma unroll
      for (int r = 0; r < 8; ++r) {
        const int row = r + 8 * hb;
        if (row < TV_HW && n < TV_HW) op[row * TV_HW + n] = s[m].p[r];
      }
    }
  }
}

extern "C" void kernel_launch(void* const* d_in, const int* in_sizes, int n_in,
                              void* d_out, int out_size, void* d_ws, size_t ws_size,
                              hipStream_t stream) {
  (void)n_in; (void)d_ws; (void)ws_size; (void)out_size;
  const float* X = (const float*)d_in[0];
  float* OUT = (float*)d_out;
  const int nmaps = in_sizes[0] / (TV_HW * TV_HW);  // B = 131072

  const int threads = 256;  // 8 waves/block, 2 maps/wave -> 16 maps/block
  const int maps_per_block = (threads / 32) * 2;
  const int blocks = (nmaps + maps_per_block - 1) / maps_per_block;
  tv2d_prox_kernel<<<blocks, threads, 0, stream>>>(X, OUT, nmaps);
}
